// ori_Propagation_Pooling_Module_13314398618078
// MI455X (gfx1250) — matrix operand order repaired and checked by reading, not compiled, not run on hardware
//
#include <hip/hip_runtime.h>
#include <hip/hip_bf16.h>

// ---------------------------------------------------------------------------
// Problem constants (reference: B=8, C=512, H=W=48)
// ---------------------------------------------------------------------------
constexpr int BB    = 8;          // batch
constexpr int CC    = 512;        // channels
constexpr int C2    = 1024;       // 2*C
constexpr int HH    = 48;
constexpr int WW    = 48;
constexpr int HWD   = 2304;       // H*W
constexpr int OHD   = 24;         // stride-2 conv output H (=W)
constexpr int OND   = 576;        // 24*24
constexpr int KCONV = 4608;       // C*3*3
constexpr int DD    = 64;         // C/8

typedef __bf16 bf16_t;
typedef __attribute__((ext_vector_type(16))) __bf16 v16bf;
typedef __attribute__((ext_vector_type(8)))  __bf16 v8bf;
typedef __attribute__((ext_vector_type(8)))  float  v8f;

__device__ __forceinline__ bf16_t f2bf(float f) {
  // round-to-nearest-even f32 -> bf16
  unsigned u = __builtin_bit_cast(unsigned, f);
  u += 0x7FFFu + ((u >> 16) & 1u);
  unsigned short s = (unsigned short)(u >> 16);
  return __builtin_bit_cast(bf16_t, s);
}

// ---------------------------------------------------------------------------
// Generic NT GEMM:  C[b, m, n] = sum_k A[b, m, k] * Bt[b, n, k]  (+ bias[m])
// A: M x K row-major (bf16), Bt: N x K row-major (bf16), K % 32 == 0.
// Block: 256 threads = 8 waves, 128x128 tile. Wave grid 2(M) x 4(N); each
// wave owns a 64x32 sub-tile = 4x2 WMMA 16x16 accumulators.
// Fragment layouts: k = 16*half + s, not the ISA 7.12.2 table order (A 16x32 and B both packed as
// contiguous K-runs per lane when sourced from an N x K row-major operand).
// ---------------------------------------------------------------------------
__global__ __launch_bounds__(256) void gemm_nt_bf16(
    const bf16_t* __restrict__ A,  long long strideA,
    const bf16_t* __restrict__ Bt, long long strideB,
    float*        __restrict__ Cf, long long strideCf,
    bf16_t*       __restrict__ Cb, long long strideCb,
    const float*  __restrict__ bias,
    int M, int N, int K, int transCb)
{
  const int lane = threadIdx.x & 31;
  const int wv   = threadIdx.x >> 5;
  const int wm   = (wv >> 2) * 64;   // 0 or 64
  const int wn   = (wv & 3) * 32;    // 0..96
  const int bm   = blockIdx.y * 128;
  const int bn   = blockIdx.x * 128;
  const int bb   = blockIdx.z;

  const bf16_t* Ab = A  + (size_t)bb * strideA;
  const bf16_t* Bb = Bt + (size_t)bb * strideB;

  const int l15 = lane & 15;
  const int hi  = (lane >> 4) & 1;   // lane half selects K sub-range

  size_t aoff[4];
  size_t boff[2];
#pragma unroll
  for (int i = 0; i < 4; ++i) {
    int m = bm + wm + i * 16 + l15;
    if (m > M - 1) m = M - 1;                 // clamp; OOB rows masked at store
    aoff[i] = (size_t)m * K + (size_t)(hi * 16);
  }
#pragma unroll
  for (int j = 0; j < 2; ++j) {
    int n = bn + wn + j * 16 + l15;
    if (n > N - 1) n = N - 1;
    boff[j] = (size_t)n * K + (size_t)(hi * 16);
  }

  v8f acc[4][2];
#pragma unroll
  for (int i = 0; i < 4; ++i)
#pragma unroll
    for (int j = 0; j < 2; ++j)
      acc[i][j] = (v8f)(0.0f);

  for (int k0 = 0; k0 < K; k0 += 32) {
    v16bf afrag[4], bfrag[2];
#pragma unroll
    for (int i = 0; i < 4; ++i) {
      const bf16_t* p = Ab + aoff[i] + k0;
      v8bf lo = *(const v8bf*)(p);        // K = kb+0..7
      v8bf hv = *(const v8bf*)(p + 8);    // K = kb+8..15
      afrag[i] = __builtin_shufflevector(lo, hv,
          0,1,2,3,4,5,6,7,8,9,10,11,12,13,14,15);
    }
#pragma unroll
    for (int j = 0; j < 2; ++j)
      bfrag[j] = *(const v16bf*)(Bb + boff[j] + k0);  // contiguous 16 K values

    if (k0 + 32 < K) {                     // global_prefetch_b8 next K panel
      __builtin_prefetch(Ab + aoff[0] + k0 + 32, 0, 1);
      __builtin_prefetch(Bb + boff[0] + k0 + 32, 0, 1);
    }

#pragma unroll
    for (int i = 0; i < 4; ++i)
#pragma unroll
      for (int j = 0; j < 2; ++j)
        acc[i][j] = __builtin_amdgcn_wmma_f32_16x16x32_bf16(
            false, afrag[i], false, bfrag[j], (short)0, acc[i][j],
            false, false);
  }

  float*  Cfb = Cf ? Cf + (size_t)bb * strideCf : nullptr;
  bf16_t* Cbb = Cb ? Cb + (size_t)bb * strideCb : nullptr;
  const int cn  = l15;        // N = lane & 15
  const int cm0 = hi * 8;     // M base per lane half (ISA 7.12.2 C layout)
#pragma unroll
  for (int i = 0; i < 4; ++i) {
#pragma unroll
    for (int j = 0; j < 2; ++j) {
      const int mt = bm + wm + i * 16;
      const int nt = bn + wn + j * 16;
#pragma unroll
      for (int e = 0; e < 8; ++e) {
        const int m = mt + cm0 + e;
        const int n = nt + cn;
        if (m < M && n < N) {
          float v = acc[i][j][e];
          if (bias) v += bias[m];
          if (Cfb) Cfb[(size_t)m * N + n] = v;
          if (Cbb) {
            if (transCb) Cbb[(size_t)n * M + m] = f2bf(v);
            else         Cbb[(size_t)m * N + n] = f2bf(v);
          }
        }
      }
    }
  }
}

// ---------------------------------------------------------------------------
// f32 -> bf16 elementwise (weight packing)
// ---------------------------------------------------------------------------
__global__ __launch_bounds__(256) void f32_to_bf16(
    const float* __restrict__ in, bf16_t* __restrict__ outp, long long n)
{
  long long i = (long long)blockIdx.x * 256 + threadIdx.x;
  if (i < n) outp[i] = f2bf(in[i]);
}

// xT[b, n, c] = x[b, c, n]  (bf16) -- N x K operand for all GEMMs against xf
__global__ __launch_bounds__(256) void pack_xT(
    const float* __restrict__ x, bf16_t* __restrict__ xT)
{
  long long idx = (long long)blockIdx.x * 256 + threadIdx.x;
  long long total = (long long)BB * HWD * CC;
  if (idx >= total) return;
  int c = (int)(idx % CC);
  int n = (int)((idx / CC) % HWD);
  int b = (int)(idx / ((long long)CC * HWD));
  xT[idx] = f2bf(x[((size_t)b * CC + c) * HWD + n]);
}

// im2col for stride-2, pad-1, 3x3 conv: col[b, n, k], k = ic*9 + kh*3 + kw
__global__ __launch_bounds__(256) void im2col_s2(
    const float* __restrict__ x, bf16_t* __restrict__ col)
{
  long long idx = (long long)blockIdx.x * 256 + threadIdx.x;
  long long total = (long long)BB * OND * KCONV;
  if (idx >= total) return;
  int k  = (int)(idx % KCONV);
  int n  = (int)((idx / KCONV) % OND);
  int b  = (int)(idx / ((long long)KCONV * OND));
  int kw = k % 3;
  int kh = (k / 3) % 3;
  int ic = k / 9;
  int ow = n % OHD;
  int oh = n / OHD;
  int ih = oh * 2 - 1 + kh;
  int iw = ow * 2 - 1 + kw;
  float v = 0.0f;
  if (ih >= 0 && ih < HH && iw >= 0 && iw < WW)
    v = x[((size_t)b * CC + ic) * HWD + ih * WW + iw];
  col[idx] = f2bf(v);
}

// Row softmax over `cols` of sign*E[row, :], bf16 output.
// sign=+1: standard softmax. sign=-1: softmax(max-E) == softmax(-E) (CAM trick).
__global__ __launch_bounds__(256) void row_softmax(
    const float* __restrict__ E, bf16_t* __restrict__ O, int cols, float sign)
{
  __shared__ float red[256];
  const int row = blockIdx.x;
  const int t   = threadIdx.x;
  const float* e = E + (size_t)row * cols;
  bf16_t*      o = O + (size_t)row * cols;

  float mx = -3.0e38f;
  for (int c = t; c < cols; c += 256) mx = fmaxf(mx, sign * e[c]);
  red[t] = mx; __syncthreads();
  for (int s = 128; s > 0; s >>= 1) {
    if (t < s) red[t] = fmaxf(red[t], red[t + s]);
    __syncthreads();
  }
  mx = red[0]; __syncthreads();

  float sum = 0.0f;
  for (int c = t; c < cols; c += 256) sum += __expf(sign * e[c] - mx);
  red[t] = sum; __syncthreads();
  for (int s = 128; s > 0; s >>= 1) {
    if (t < s) red[t] += red[t + s];
    __syncthreads();
  }
  const float inv = 1.0f / red[0];

  for (int c = t; c < cols; c += 256)
    o[c] = f2bf(__expf(sign * e[c] - mx) * inv);
}

// ---------------------------------------------------------------------------
// Launcher
// ---------------------------------------------------------------------------
extern "C" void kernel_launch(void* const* d_in, const int* in_sizes, int n_in,
                              void* d_out, int out_size, void* d_ws, size_t ws_size,
                              hipStream_t stream) {
  (void)in_sizes; (void)n_in; (void)out_size; (void)ws_size;
  const float* x   = (const float*)d_in[0];
  const float* Wqc = (const float*)d_in[1];
  const float* bqc = (const float*)d_in[2];
  const float* Wkc = (const float*)d_in[3];
  const float* bkc = (const float*)d_in[4];
  const float* Wqp = (const float*)d_in[5];
  const float* bqp = (const float*)d_in[6];
  const float* Wkp = (const float*)d_in[7];
  const float* bkp = (const float*)d_in[8];
  float* out = (float*)d_out;

  char* wsp = (char*)d_ws;
  auto take = [&](size_t bytes) -> char* {
    char* p = wsp;
    wsp += (bytes + 255) & ~(size_t)255;
    return p;
  };

  bf16_t* xT     = (bf16_t*)take((size_t)BB * HWD * CC * 2);    // (b, n, c)
  bf16_t* col    = (bf16_t*)take((size_t)BB * OND * KCONV * 2); // (b, n, k)
  bf16_t* wqc_b  = (bf16_t*)take((size_t)C2 * KCONV * 2);
  bf16_t* wkc_b  = (bf16_t*)take((size_t)CC * KCONV * 2);
  bf16_t* wqp_b  = (bf16_t*)take((size_t)DD * CC * 2);
  bf16_t* wkp_b  = (bf16_t*)take((size_t)DD * CC * 2);
  bf16_t* qc_b   = (bf16_t*)take((size_t)BB * C2 * OND * 2);    // (b, 2C, 576)
  bf16_t* kc_b   = (bf16_t*)take((size_t)BB * CC * OND * 2);    // (b, C, 576)
  float*  energy = (float*) take((size_t)BB * C2 * CC * 4);     // (b, 2C, C)
  bf16_t* attnc  = (bf16_t*)take((size_t)BB * C2 * CC * 2);
  bf16_t* outcb  = (bf16_t*)take((size_t)BB * C2 * HWD * 2);    // (b, 2C, HW)
  bf16_t* qpT    = (bf16_t*)take((size_t)BB * HWD * DD * 2);    // (b, HW, 64)
  bf16_t* kpT    = (bf16_t*)take((size_t)BB * HWD * DD * 2);
  float*  ep     = (float*) take((size_t)HWD * HWD * 4);        // per-batch reuse
  bf16_t* attnp  = (bf16_t*)take((size_t)HWD * HWD * 2);        // per-batch reuse

  auto cvt = [&](const float* src, bf16_t* dst, long long n) {
    f32_to_bf16<<<(unsigned)((n + 255) / 256), 256, 0, stream>>>(src, dst, n);
  };
  cvt(Wqc, wqc_b, (long long)C2 * KCONV);
  cvt(Wkc, wkc_b, (long long)CC * KCONV);
  cvt(Wqp, wqp_b, (long long)DD * CC);
  cvt(Wkp, wkp_b, (long long)DD * CC);

  {
    long long n = (long long)BB * HWD * CC;
    pack_xT<<<(unsigned)((n + 255) / 256), 256, 0, stream>>>(x, xT);
  }
  {
    long long n = (long long)BB * OND * KCONV;
    im2col_s2<<<(unsigned)((n + 255) / 256), 256, 0, stream>>>(x, col);
  }

  auto gemm = [&](const bf16_t* A, long long sA, const bf16_t* Bt, long long sB,
                  float* Cf, long long sCf, bf16_t* Cb, long long sCb,
                  const float* bias, int M, int N, int K, int trans, int batch) {
    dim3 g((unsigned)((N + 127) / 128), (unsigned)((M + 127) / 128), (unsigned)batch);
    gemm_nt_bf16<<<g, 256, 0, stream>>>(A, sA, Bt, sB, Cf, sCf, Cb, sCb, bias,
                                        M, N, K, trans);
  };

  // CAM q/k convs as implicit GEMM over im2col panels (weights batch-shared).
  gemm(wqc_b, 0, col, (long long)OND * KCONV, nullptr, 0,
       qc_b, (long long)C2 * OND, bqc, C2, OND, KCONV, 0, BB);
  gemm(wkc_b, 0, col, (long long)OND * KCONV, nullptr, 0,
       kc_b, (long long)CC * OND, bkc, CC, OND, KCONV, 0, BB);

  // energy[b,q,c] = sum_m qc[b,q,m] * kc[b,c,m]  (NT: Bt = kc as stored)
  gemm(qc_b, (long long)C2 * OND, kc_b, (long long)CC * OND,
       energy, (long long)C2 * CC, nullptr, 0, nullptr, C2, CC, OND, 0, BB);

  // softmax(max - energy) == softmax(-energy): sign = -1
  row_softmax<<<BB * C2, 256, 0, stream>>>(energy, attnc, CC, -1.0f);

  // out_c = attn_c @ xf  -> f32 to d_out[0] AND bf16 copy for the PAM GEMM
  gemm(attnc, (long long)C2 * CC, xT, (long long)HWD * CC,
       out, (long long)C2 * HWD, outcb, (long long)C2 * HWD,
       nullptr, C2, HWD, CC, 0, BB);

  // qp/kp 1x1 convs; epilogue writes transposed (HW x 64) operands
  gemm(wqp_b, 0, xT, (long long)HWD * CC, nullptr, 0,
       qpT, (long long)HWD * DD, bqp, DD, HWD, CC, 1, BB);
  gemm(wkp_b, 0, xT, (long long)HWD * CC, nullptr, 0,
       kpT, (long long)HWD * DD, bkp, DD, HWD, CC, 1, BB);

  // PAM per batch (stream-serialized) so the 2304x2304 buffers are reused.
  for (int b = 0; b < BB; ++b) {
    // energy_p[n,m] = sum_d qpT[n,d] * kpT[m,d]
    gemm(qpT + (size_t)b * HWD * DD, 0, kpT + (size_t)b * HWD * DD, 0,
         ep, 0, nullptr, 0, nullptr, HWD, HWD, DD, 0, 1);
    row_softmax<<<HWD, 256, 0, stream>>>(ep, attnp, HWD, 1.0f);
    // out_p[q,m] = sum_n out_c[q,n] * attn_p[m,n]  (NT: Bt = attn_p rows)
    gemm(outcb + (size_t)b * C2 * HWD, 0, attnp, 0,
         out + (size_t)BB * C2 * HWD + (size_t)b * C2 * HWD, 0,
         nullptr, 0, nullptr, C2, HWD, HWD, 0, 1);
  }
}
